// ID_50294067036307
// MI455X (gfx1250) — matrix operand order repaired and checked by reading, not compiled, not run on hardware
//
#include <hip/hip_runtime.h>
#include <hip/hip_bf16.h>

#define N_B 64
#define C_D 512
#define L_D 256
#define EPS_D2 1e-12f

typedef __attribute__((ext_vector_type(16))) __bf16 v16bf;
typedef __attribute__((ext_vector_type(8)))  __bf16 v8bf;
typedef __attribute__((ext_vector_type(8)))  float  v8f;

// ---------------- workspace layout (bytes) ----------------
// EShi | ESlo | EThi | ETlo : N*C*L bf16 each
// sqS | sqT                 : N*C f32
// sums_s | sums_t           : N f32
// Ds | Dt                   : N*C*C f32
static constexpr size_t O_ESHI = 0;
static constexpr size_t SZ_E   = (size_t)N_B * C_D * L_D * sizeof(__bf16);   // 16 MiB
static constexpr size_t O_ESLO = O_ESHI + SZ_E;
static constexpr size_t O_ETHI = O_ESLO + SZ_E;
static constexpr size_t O_ETLO = O_ETHI + SZ_E;
static constexpr size_t O_SQS  = O_ETLO + SZ_E;
static constexpr size_t SZ_SQ  = (size_t)N_B * C_D * sizeof(float);
static constexpr size_t O_SQT  = O_SQS + SZ_SQ;
static constexpr size_t O_SUMS = O_SQT + SZ_SQ;
static constexpr size_t O_SUMT = O_SUMS + N_B * sizeof(float);
static constexpr size_t O_DS   = O_SUMT + N_B * sizeof(float);
static constexpr size_t SZ_D   = (size_t)N_B * C_D * C_D * sizeof(float);    // 64 MiB
static constexpr size_t O_DT   = O_DS + SZ_D;

__device__ __forceinline__ v8bf ld8(const __bf16* p) {
    return *(const v8bf*)p;   // 16B aligned by construction
}
__device__ __forceinline__ v16bf cat16(v8bf a, v8bf b) {
    return __builtin_shufflevector(a, b, 0,1,2,3,4,5,6,7,8,9,10,11,12,13,14,15);
}

// ---------------- init: zero accumulators + output ----------------
__global__ void ID_50294067036307_init(float* sums_s, float* sums_t, float* out) {
    int t = threadIdx.x;
    if (t < N_B) { sums_s[t] = 0.f; sums_t[t] = 0.f; }
    if (t == 0)  { out[0] = 0.f; }
}

// ---------------- phase P: mask, bf16 hi/lo split, exact fp32 row norms ----------------
__global__ __launch_bounds__(256) void ID_50294067036307_prep(
    const float* __restrict__ les, const float* __restrict__ let,
    const int* __restrict__ tgt,
    __bf16* __restrict__ eshi, __bf16* __restrict__ eslo,
    __bf16* __restrict__ ethi, __bf16* __restrict__ etlo,
    float* __restrict__ sqS, float* __restrict__ sqT)
{
    const int row = blockIdx.x;            // n*C + c
    const int l   = threadIdx.x;           // 0..255 == L
    const float m = (float)tgt[row];
    const size_t idx = (size_t)row * L_D + l;

    float xs = les[idx] * m;
    float xt = let[idx] * m;

    __bf16 hs = (__bf16)xs;  __bf16 ls = (__bf16)(xs - (float)hs);
    __bf16 ht = (__bf16)xt;  __bf16 lt = (__bf16)(xt - (float)ht);
    eshi[idx] = hs; eslo[idx] = ls;
    ethi[idx] = ht; etlo[idx] = lt;

    __shared__ float bs[256], bt[256];
    bs[l] = xs * xs; bt[l] = xt * xt;
    __syncthreads();
    for (int off = 128; off > 0; off >>= 1) {
        if (l < off) { bs[l] += bs[l + off]; bt[l] += bt[l + off]; }
        __syncthreads();
    }
    if (l == 0) { sqS[row] = bs[0]; sqT[row] = bt[0]; }
}

// ---------------- compensated bf16 Gram tile: 16x16 output, K = 256 ----------------
__device__ __forceinline__ v8f gram16x16(const __bf16* __restrict__ Hi,
                                         const __bf16* __restrict__ Lo,
                                         int i0, int j0, int lane)
{
    v8f acc = {};
    const int lsub = lane & 15;
    const __bf16* arhi = Hi + (size_t)(i0 + lsub) * L_D;
    const __bf16* arlo = Lo + (size_t)(i0 + lsub) * L_D;
    const __bf16* brhi = Hi + (size_t)(j0 + lsub) * L_D;
    const __bf16* brlo = Lo + (size_t)(j0 + lsub) * L_D;
    const int as0 = (lane < 16) ? 0 : 8;    // A fragment K-segments per lane
    const int bs0 = (lane < 16) ? 0 : 8;    // B fragment K-segments per lane

    #pragma unroll
    for (int kb = 0; kb < L_D; kb += 32) {
        if (kb + 32 < L_D) {                 // prefetch next K chunk
            __builtin_prefetch(arhi + kb + 32, 0, 1);
            __builtin_prefetch(brhi + kb + 32, 0, 1);
        }
        v16bf ahi = cat16(ld8(arhi + kb + as0), ld8(arhi + kb + 16 + as0));
        v16bf alo = cat16(ld8(arlo + kb + as0), ld8(arlo + kb + 16 + as0));
        v16bf bhi = cat16(ld8(brhi + kb + bs0), ld8(brhi + kb + 16 + bs0));
        v16bf blo = cat16(ld8(brlo + kb + bs0), ld8(brlo + kb + 16 + bs0));

        acc = __builtin_amdgcn_wmma_f32_16x16x32_bf16(false, ahi, false, bhi,
                                                      (short)0, acc, false, false);
        acc = __builtin_amdgcn_wmma_f32_16x16x32_bf16(false, ahi, false, blo,
                                                      (short)0, acc, false, false);
        acc = __builtin_amdgcn_wmma_f32_16x16x32_bf16(false, alo, false, bhi,
                                                      (short)0, acc, false, false);
    }
    return acc;
}

// ---------------- phase A: masked distance tiles + mean numerators ----------------
__global__ __launch_bounds__(256) void ID_50294067036307_dist(
    const __bf16* __restrict__ eshi, const __bf16* __restrict__ eslo,
    const __bf16* __restrict__ ethi, const __bf16* __restrict__ etlo,
    const float* __restrict__ sqS, const float* __restrict__ sqT,
    const int* __restrict__ tgt,
    float* __restrict__ Ds, float* __restrict__ Dt,
    float* __restrict__ sums_s, float* __restrict__ sums_t)
{
    const int n   = blockIdx.z;
    const int bi  = blockIdx.y;              // 0..7  (64-row strip)
    const int bj  = blockIdx.x;              // 0..15 (32-col strip)
    const int tid = threadIdx.x;
    const int lane = tid & 31;
    const int w    = tid >> 5;               // 8 waves: 4x2 subtiles
    const int i0 = bi * 64 + (w >> 1) * 16;
    const int j0 = bj * 32 + (w & 1) * 16;

    const size_t eoff = (size_t)n * C_D * L_D;
    const int    nc   = n * C_D;
    const size_t doff = (size_t)n * C_D * C_D;

    const int j  = j0 + (lane & 15);
    const float mj  = (float)tgt[nc + j];
    const float sjS = sqS[nc + j];
    const float sjT = sqT[nc + j];
    const int rbase = i0 + ((lane >> 4) << 3);   // rows r..r+7 for this half-wave

    float psS = 0.f, psT = 0.f;

    // ---- student ----
    {
        v8f g = gram16x16(eshi + eoff, eslo + eoff, i0, j0, lane);
        #pragma unroll
        for (int r = 0; r < 8; ++r) {
            int i = rbase + r;
            float mi = (float)tgt[nc + i];
            float d2 = sqS[nc + i] + sjS - 2.f * g[r];
            float d  = sqrtf(fmaxf(d2, EPS_D2));
            float v  = (i == j) ? 0.f : d * mi * mj;
            Ds[doff + (size_t)i * C_D + j] = v;
            psS += v;
        }
    }
    // ---- teacher ----
    {
        v8f g = gram16x16(ethi + eoff, etlo + eoff, i0, j0, lane);
        #pragma unroll
        for (int r = 0; r < 8; ++r) {
            int i = rbase + r;
            float mi = (float)tgt[nc + i];
            float d2 = sqT[nc + i] + sjT - 2.f * g[r];
            float d  = sqrtf(fmaxf(d2, EPS_D2));
            float v  = (i == j) ? 0.f : d * mi * mj;
            Dt[doff + (size_t)i * C_D + j] = v;
            psT += v;
        }
    }

    __shared__ float bs[256], bt[256];
    bs[tid] = psS; bt[tid] = psT;
    __syncthreads();
    for (int off = 128; off > 0; off >>= 1) {
        if (tid < off) { bs[tid] += bs[tid + off]; bt[tid] += bt[tid + off]; }
        __syncthreads();
    }
    if (tid == 0) {
        atomicAdd(&sums_s[n], bs[0]);
        atomicAdd(&sums_t[n], bt[0]);
    }
}

// ---------------- phase B: normalize + Huber + scalar reduce ----------------
__global__ __launch_bounds__(256) void ID_50294067036307_huber(
    const float* __restrict__ Ds, const float* __restrict__ Dt,
    const float* __restrict__ sums_s, const float* __restrict__ sums_t,
    const int* __restrict__ tgt, float* __restrict__ out)
{
    const int n = blockIdx.x;
    const int tid = threadIdx.x;
    __shared__ float buf[256];
    __shared__ float s_inv_ms, s_inv_mt, s_npos;

    float np = 0.f;
    for (int c = tid; c < C_D; c += 256) np += (float)tgt[n * C_D + c];
    buf[tid] = np;
    __syncthreads();
    for (int off = 128; off > 0; off >>= 1) {
        if (tid < off) buf[tid] += buf[tid + off];
        __syncthreads();
    }
    if (tid == 0) {
        float npos = buf[0];
        float cnt  = npos * (npos - 1.f);
        float safe = fmaxf(cnt, 1.f);
        bool valid = npos > 1.f;
        float ms = valid ? (sums_s[n] / safe) : 1.f;
        float mt = valid ? (sums_t[n] / safe) : 1.f;
        s_inv_ms = 1.f / ms;
        s_inv_mt = 1.f / mt;
        s_npos   = npos;
    }
    __syncthreads();
    const float inv_ms = s_inv_ms, inv_mt = s_inv_mt;
    const size_t doff = (size_t)n * C_D * C_D;

    float h = 0.f;
    for (int idx = tid; idx < C_D * C_D; idx += 256) {
        float diff = Ds[doff + idx] * inv_ms - Dt[doff + idx] * inv_mt;
        float a = fabsf(diff);
        h += (a < 1.f) ? 0.5f * diff * diff : a - 0.5f;
    }
    buf[tid] = h;
    __syncthreads();
    for (int off = 128; off > 0; off >>= 1) {
        if (tid < off) buf[tid] += buf[tid + off];
        __syncthreads();
    }
    if (tid == 0 && s_npos > 1.f) {
        atomicAdd(out, buf[0] / fmaxf(s_npos, 1.f));
    }
}

extern "C" void kernel_launch(void* const* d_in, const int* in_sizes, int n_in,
                              void* d_out, int out_size, void* d_ws, size_t ws_size,
                              hipStream_t stream) {
    const float* les = (const float*)d_in[0];
    const float* let = (const float*)d_in[1];
    const int*   tgt = (const int*)d_in[2];
    float* out = (float*)d_out;

    char* ws = (char*)d_ws;
    __bf16* eshi = (__bf16*)(ws + O_ESHI);
    __bf16* eslo = (__bf16*)(ws + O_ESLO);
    __bf16* ethi = (__bf16*)(ws + O_ETHI);
    __bf16* etlo = (__bf16*)(ws + O_ETLO);
    float* sqS    = (float*)(ws + O_SQS);
    float* sqT    = (float*)(ws + O_SQT);
    float* sums_s = (float*)(ws + O_SUMS);
    float* sums_t = (float*)(ws + O_SUMT);
    float* Ds     = (float*)(ws + O_DS);
    float* Dt     = (float*)(ws + O_DT);

    ID_50294067036307_init<<<1, 256, 0, stream>>>(sums_s, sums_t, out);

    ID_50294067036307_prep<<<N_B * C_D, L_D, 0, stream>>>(
        les, let, tgt, eshi, eslo, ethi, etlo, sqS, sqT);

    dim3 gA(C_D / 32, C_D / 64, N_B);   // (16, 8, 64)
    ID_50294067036307_dist<<<gA, 256, 0, stream>>>(
        eshi, eslo, ethi, etlo, sqS, sqT, tgt, Ds, Dt, sums_s, sums_t);

    ID_50294067036307_huber<<<N_B, 256, 0, stream>>>(
        Ds, Dt, sums_s, sums_t, tgt, out);
}
